// RNNFC_57174604644868
// MI455X (gfx1250) — matrix operand order repaired and checked by reading, not compiled, not run on hardware
//
#include <hip/hip_runtime.h>

// ---------------------------------------------------------------------------
// Problem dims (compile-time)
// ---------------------------------------------------------------------------
#define BSZ   64
#define TT    512
#define IND   512
#define HIDD  1024
#define OUTD  512
#define MROWS (BSZ * TT)   // 32768 flattened (b,t) rows for the x-projection
#define NBLK  16           // persistent-kernel grid size (co-resident, barrier-safe)
#define WLDS  (HIDD + 8)   // padded LDS stride (bank-conflict-free, 16B aligned)

typedef __bf16 bf16;
typedef bf16  v16bf __attribute__((ext_vector_type(16)));
typedef bf16  v8bf  __attribute__((ext_vector_type(8)));
typedef float v8f   __attribute__((ext_vector_type(8)));

#define WMMA_BF16(A, B, C) \
    __builtin_amdgcn_wmma_f32_16x16x32_bf16(false, (A), false, (B), (short)0, (C), false, false)

// ---------------------------------------------------------------------------
// fp32 -> bf16 (round-to-nearest-even) without bf16 arithmetic
// ---------------------------------------------------------------------------
__device__ __forceinline__ unsigned short f2bf(float x) {
    unsigned int u = __float_as_uint(x);
    u += 0x7FFFu + ((u >> 16) & 1u);
    return (unsigned short)(u >> 16);
}

__global__ void cvt_bf16(const float* __restrict__ src,
                         unsigned short* __restrict__ dst, int n) {
    int i = blockIdx.x * blockDim.x + threadIdx.x;
    int stride = gridDim.x * blockDim.x;
    for (; i < n; i += stride) dst[i] = f2bf(src[i]);
}

__global__ void init_state(unsigned short* __restrict__ h0, int n,
                           unsigned int* __restrict__ barrier_cnt) {
    int i = blockIdx.x * blockDim.x + threadIdx.x;
    int stride = gridDim.x * blockDim.x;
    for (; i < n; i += stride) h0[i] = 0;     // bf16 zero
    if (blockIdx.x == 0 && threadIdx.x == 0) *barrier_cnt = 0;
}

// ---------------------------------------------------------------------------
// WMMA fragment loaders (bf16, NT layout: A[M,K] row-major, W[N,K] row-major)
//
// A 16x32 bf16 fragment (contiguous k order chosen here, not the ISA table order; B matches):
//   lanes 0-15 : row M=lane,    elems 0..15 = K k0+0..15
//   lanes 16-31: row M=lane-16, elems 0..15 = K k0+16..31
// B 32x16 bf16 fragment: B[k][n] = W[n][k]
//   lanes 0-15 : col N=lane,    elems 0..15 = K k0+0..15
//   lanes 16-31: col N=lane-16, elems 0..15 = K k0+16..31
// ---------------------------------------------------------------------------
__device__ __forceinline__ v16bf load_a_frag(const unsigned short* __restrict__ A,
                                             int row0, int lda, int k0) {
    int lane = threadIdx.x & 31;
    const bf16* p = reinterpret_cast<const bf16*>(A)
                  + (size_t)(row0 + (lane & 15)) * lda
                  + k0 + ((lane >> 4) << 4);
    v8bf lo = *reinterpret_cast<const v8bf*>(p);
    v8bf hi = *reinterpret_cast<const v8bf*>(p + 8);
    return __builtin_shufflevector(lo, hi, 0, 1, 2, 3, 4, 5, 6, 7,
                                           8, 9, 10, 11, 12, 13, 14, 15);
}

__device__ __forceinline__ v16bf load_b_frag(const unsigned short* __restrict__ W,
                                             int col0, int ldw, int k0) {
    int lane = threadIdx.x & 31;
    const bf16* p = reinterpret_cast<const bf16*>(W)
                  + (size_t)(col0 + (lane & 15)) * ldw
                  + k0 + ((lane >> 4) << 4);
    return *(const v16bf*)(p);
}

// B-fragment sourced from LDS (two 16B ds_load_b128's; stride WLDS keeps
// lanes 0-15 on disjoint bank quads and every access 16B aligned).
__device__ __forceinline__ v16bf load_b_frag_lds(const unsigned short* Wl,
                                                 int col0, int k0) {
    int lane = threadIdx.x & 31;
    const unsigned short* p = Wl + (size_t)(col0 + (lane & 15)) * WLDS
                                 + k0 + ((lane >> 4) << 4);
    v8bf lo = *reinterpret_cast<const v8bf*>(p);
    v8bf hi = *reinterpret_cast<const v8bf*>(p + 8);
    return __builtin_shufflevector(lo, hi, 0, 1, 2, 3, 4, 5, 6, 7,
                                           8, 9, 10, 11, 12, 13, 14, 15);
}

// ---------------------------------------------------------------------------
// Phase 1: xp[b*T+t, n] = input @ W_ih^T + b   (M=32768, N=1024, K=512)
// Block: 256 threads = 8 waves (2x4), wave tile 32x32, block tile 64x128.
// Grid: (N/128, M/64) = (8, 512)
// ---------------------------------------------------------------------------
__global__ __launch_bounds__(256) void xproj_gemm(
        const unsigned short* __restrict__ Ab,   // bf16 [32768,512]
        const unsigned short* __restrict__ Wb,   // bf16 [1024,512]
        const float* __restrict__ bias,          // [1024]
        float* __restrict__ xp) {                // [32768,1024]
    const int K = IND, N = HIDD;
    int w  = threadIdx.x >> 5;
    int wr = w >> 2, wc = w & 3;
    int row0 = blockIdx.y * 64 + wr * 32;
    int col0 = blockIdx.x * 128 + wc * 32;

    v8f acc[2][2] = {};
    for (int k = 0; k < K; k += 32) {
        v16bf a0 = load_a_frag(Ab, row0,      K, k);
        v16bf a1 = load_a_frag(Ab, row0 + 16, K, k);
        v16bf b0 = load_b_frag(Wb, col0,      K, k);
        v16bf b1 = load_b_frag(Wb, col0 + 16, K, k);
        acc[0][0] = WMMA_BF16(a0, b0, acc[0][0]);
        acc[0][1] = WMMA_BF16(a0, b1, acc[0][1]);
        acc[1][0] = WMMA_BF16(a1, b0, acc[1][0]);
        acc[1][1] = WMMA_BF16(a1, b1, acc[1][1]);
    }

    int lane = threadIdx.x & 31;
    int ln = lane & 15, lm = (lane >> 4) << 3;
#pragma unroll
    for (int i = 0; i < 2; ++i)
#pragma unroll
        for (int j = 0; j < 2; ++j) {
            int cm = row0 + i * 16 + lm;
            int cn = col0 + j * 16 + ln;
            float bv = bias[cn];
#pragma unroll
            for (int r = 0; r < 8; ++r)
                xp[(size_t)(cm + r) * N + cn] = acc[i][j][r] + bv;
        }
}

// ---------------------------------------------------------------------------
// Grid-wide barrier for the persistent kernel (monotonic counter, no reset).
// ---------------------------------------------------------------------------
__device__ __forceinline__ void grid_barrier(unsigned int* cnt, unsigned int target) {
    __threadfence();                 // release all prior global writes
    __syncthreads();
    if (threadIdx.x == 0) {
        __hip_atomic_fetch_add(cnt, 1u, __ATOMIC_RELEASE, __HIP_MEMORY_SCOPE_AGENT);
        while (__hip_atomic_load(cnt, __ATOMIC_ACQUIRE, __HIP_MEMORY_SCOPE_AGENT) < target)
            __builtin_amdgcn_s_sleep(2);
    }
    __syncthreads();
    __threadfence();                 // acquire: drop stale cached lines
}

// ---------------------------------------------------------------------------
// Persistent recurrence kernel: all T=512 steps in one launch.
// Grid: NBLK=16 blocks x 256 threads (8 waves).
//
// Startup: block stages its 64-col W_hh slice (129 KB) AND its 32-col W_out
// slice (66 KB) into LDS (total ~193.5 KB of the WGP's 320 KB) -- all weight
// traffic for all 512 steps is LDS-resident afterwards.
// Per step t:
//   h-phase : block owns 64x64 tile of h_new = tanh(xp_t + h@W_hh^T)
//             (waves 2x4, wave tile 32x16, B-frags from LDS, K=1024)
//   barrier : one grid barrier (ping-pong + barrier chain covers all hazards)
//   out-phase: block owns 64x32 tile of out_t = h_new@W_out^T + b_out
//             (waves 4x2, wave tile 16x16, B-frags from LDS, K=1024)
// ---------------------------------------------------------------------------
__global__ __launch_bounds__(256) void rnn_persistent(
        unsigned short* __restrict__ h0,         // bf16 [64,1024] ping
        unsigned short* __restrict__ h1,         // bf16 [64,1024] pong
        const unsigned short* __restrict__ Whh,  // bf16 [1024,1024]
        const unsigned short* __restrict__ Wout, // bf16 [512,1024]
        const float* __restrict__ xp,            // [B,T,HID] fp32
        const float* __restrict__ bout,          // [512]
        float* __restrict__ out,                 // [B,T,OUT] fp32
        unsigned int* __restrict__ barrier_cnt) {
    __shared__ unsigned short whh_lds[64 * WLDS];   // 64 cols x (1024+8) bf16 ~129 KB
    __shared__ unsigned short wout_lds[32 * WLDS];  // 32 cols x (1024+8) bf16 ~ 66 KB

    const int lane = threadIdx.x & 31;
    const int ln = lane & 15, lm = (lane >> 4) << 3;
    const int w = threadIdx.x >> 5;

    // ---- one-time stage of this block's weight slices into LDS ----
    {
        const bf16* srcH = reinterpret_cast<const bf16*>(Whh);
        const int colH = blockIdx.x * 64;
        for (int idx = threadIdx.x; idx < 64 * (HIDD / 8); idx += 256) {
            int c  = idx >> 7;              // column 0..63  (HIDD/8 = 128 chunks)
            int kc = (idx & 127) << 3;      // element offset 0..1016 step 8
            *reinterpret_cast<v8bf*>(&whh_lds[c * WLDS + kc]) =
                *reinterpret_cast<const v8bf*>(srcH + (size_t)(colH + c) * HIDD + kc);
        }
        const bf16* srcO = reinterpret_cast<const bf16*>(Wout);
        const int colO = blockIdx.x * 32;
        for (int idx = threadIdx.x; idx < 32 * (HIDD / 8); idx += 256) {
            int c  = idx >> 7;              // column 0..31
            int kc = (idx & 127) << 3;
            *reinterpret_cast<v8bf*>(&wout_lds[c * WLDS + kc]) =
                *reinterpret_cast<const v8bf*>(srcO + (size_t)(colO + c) * HIDD + kc);
        }
    }
    __syncthreads();

    // h-phase tile coords: block covers cols [blk*64, blk*64+64)
    const int h_row0 = (w >> 2) * 32;                       // 0 or 32
    const int h_lcol = (w & 3) * 16;                        // LDS-local column
    const int h_col0 = blockIdx.x * 64 + h_lcol;            // global column

    // out-phase tile coords: block covers cols [blk*32, blk*32+32)
    const int o_row0 = (w >> 1) * 16;                       // 0,16,32,48
    const int o_lcol = (w & 1) * 16;                        // LDS-local column
    const int o_col0 = blockIdx.x * 32 + o_lcol;

    for (int t = 0; t < TT; ++t) {
        const unsigned short* hin = (t & 1) ? h1 : h0;
        unsigned short*       hout = (t & 1) ? h0 : h1;

        // ---- h-phase: h_new = tanh(xp_t + h @ W_hh^T) ----
        {
            v8f acc[2] = {};
            for (int k = 0; k < HIDD; k += 32) {
                v16bf a0 = load_a_frag(hin, h_row0,      HIDD, k);
                v16bf a1 = load_a_frag(hin, h_row0 + 16, HIDD, k);
                v16bf b0 = load_b_frag_lds(whh_lds, h_lcol, k);
                acc[0] = WMMA_BF16(a0, b0, acc[0]);
                acc[1] = WMMA_BF16(a1, b0, acc[1]);
            }
            int cn = h_col0 + ln;
#pragma unroll
            for (int i = 0; i < 2; ++i) {
                int cm = h_row0 + i * 16 + lm;
#pragma unroll
                for (int r = 0; r < 8; ++r) {
                    int m = cm + r;
                    float v = acc[i][r] + xp[((size_t)m * TT + t) * HIDD + cn];
                    hout[(size_t)m * HIDD + cn] = f2bf(tanhf(v));
                }
            }
        }

        // ---- grid barrier: all h_new tiles visible everywhere ----
        grid_barrier(barrier_cnt, (unsigned int)(t + 1) * NBLK);

        // ---- out-phase: out_t = h_new @ W_out^T + b_out ----
        {
            v8f acc = {};
            for (int k = 0; k < HIDD; k += 32) {
                v16bf a  = load_a_frag(hout, o_row0, HIDD, k);
                v16bf b0 = load_b_frag_lds(wout_lds, o_lcol, k);
                acc = WMMA_BF16(a, b0, acc);
            }
            int cn = o_col0 + ln;
            int cm = o_row0 + lm;
            float bv = bout[cn];
#pragma unroll
            for (int r = 0; r < 8; ++r) {
                int m = cm + r;
                out[((size_t)m * TT + t) * OUTD + cn] = acc[r] + bv;
            }
        }
    }
}

// ---------------------------------------------------------------------------
// Host-side launcher
// ---------------------------------------------------------------------------
extern "C" void kernel_launch(void* const* d_in, const int* in_sizes, int n_in,
                              void* d_out, int out_size, void* d_ws, size_t ws_size,
                              hipStream_t stream) {
    (void)in_sizes; (void)n_in; (void)out_size; (void)ws_size;

    const float* input = (const float*)d_in[0];   // [B,T,IN]
    const float* W_ih  = (const float*)d_in[1];   // [HID,IN]
    const float* W_hh  = (const float*)d_in[2];   // [HID,HID]
    const float* bias  = (const float*)d_in[3];   // [HID]
    const float* W_out = (const float*)d_in[4];   // [OUT,HID]
    const float* b_out = (const float*)d_in[5];   // [OUT]
    float* out = (float*)d_out;                   // [B,T,OUT]

    char* ws = (char*)d_ws;
    size_t off = 0;
    float* xp = (float*)(ws + off);                     off += (size_t)MROWS * HIDD * 4;
    unsigned short* inb  = (unsigned short*)(ws + off); off += (size_t)MROWS * IND * 2;
    unsigned short* wih  = (unsigned short*)(ws + off); off += (size_t)HIDD * IND * 2;
    unsigned short* whh  = (unsigned short*)(ws + off); off += (size_t)HIDD * HIDD * 2;
    unsigned short* wout = (unsigned short*)(ws + off); off += (size_t)OUTD * HIDD * 2;
    unsigned short* h0   = (unsigned short*)(ws + off); off += (size_t)BSZ * HIDD * 2;
    unsigned short* h1   = (unsigned short*)(ws + off); off += (size_t)BSZ * HIDD * 2;
    unsigned int* barrier_cnt = (unsigned int*)(ws + off);

    // bf16 conversions (one pass each) + state init (h0 = 0, barrier = 0)
    cvt_bf16<<<2048, 256, 0, stream>>>(input, inb, MROWS * IND);
    cvt_bf16<<<512,  256, 0, stream>>>(W_ih,  wih, HIDD * IND);
    cvt_bf16<<<1024, 256, 0, stream>>>(W_hh,  whh, HIDD * HIDD);
    cvt_bf16<<<512,  256, 0, stream>>>(W_out, wout, OUTD * HIDD);
    init_state<<<64,  256, 0, stream>>>(h0, BSZ * HIDD, barrier_cnt);

    // Phase 1: big input-projection GEMM
    dim3 gx(HIDD / 128, MROWS / 64);
    xproj_gemm<<<gx, 256, 0, stream>>>(inb, wih, bias, xp);

    // Phase 2: whole recurrence in one persistent cooperative kernel
    rnn_persistent<<<NBLK, 256, 0, stream>>>(h0, h1, whh, wout, xp, b_out,
                                             out, barrier_cnt);
}
